// ConvModule_63565515980890
// MI455X (gfx1250) — matrix operand order repaired and checked by reading, not compiled, not run on hardware
//
#include <hip/hip_runtime.h>

typedef __attribute__((ext_vector_type(16))) _Float16 v16h;
typedef __attribute__((ext_vector_type(8)))  _Float16 v8h;
typedef __attribute__((ext_vector_type(8)))  float    v8f;

namespace {
constexpr int   T  = 4, Bc = 16, C = 256, N = 1024, O = 3 * C;   // O = 768
constexpr int   M  = T * Bc * N;                                 // 65536 GEMM rows
constexpr float VTH = 1.0f, EPS = 1e-5f;
constexpr float DECAY = 0.5f;                                    // 1 - 1/TAU, TAU=2

// workspace layout (bytes)
constexpr size_t OFF_WS  = 0;                                    // softmax(w_head): C floats
constexpr size_t OFF_WPK = 1024;                                 // packed B fragments: O*C halves
constexpr size_t OFF_B1S = OFF_WPK + (size_t)O * C * 2;          // bn1 scale  (O floats)
constexpr size_t OFF_B1H = OFF_B1S + (size_t)O * 4;              // bn1 shift
constexpr size_t OFF_SUM = OFF_B1H + (size_t)O * 4;              // branch sums
constexpr size_t OFF_SQ  = OFF_SUM + (size_t)O * 4;              // branch sum-squares
constexpr size_t OFF_B2S = OFF_SQ  + (size_t)O * 4;              // bn2/3/4 scale
constexpr size_t OFF_B2H = OFF_B2S + (size_t)O * 4;              // bn2/3/4 shift
constexpr size_t OFF_H   = ((OFF_B2H + (size_t)O * 4 + 255) / 256) * 256; // spikes [T,B,N,C] f16
constexpr size_t OFF_P   = OFF_H + (size_t)M * C * 2;            // p [T,B,O,N] f32
}

// ---------------- softmax over w_head (C = 256) ----------------
__global__ void k_softmax(const float* __restrict__ w, float* __restrict__ ws) {
  __shared__ float red[C];
  int tid = threadIdx.x;
  float v = w[tid];
  red[tid] = v; __syncthreads();
  for (int s = C / 2; s > 0; s >>= 1) {
    if (tid < s) red[tid] = fmaxf(red[tid], red[tid + s]);
    __syncthreads();
  }
  float mx = red[0]; __syncthreads();
  float e = __expf(v - mx);
  red[tid] = e; __syncthreads();
  for (int s = C / 2; s > 0; s >>= 1) {
    if (tid < s) red[tid] += red[tid + s];
    __syncthreads();
  }
  ws[tid] = e / red[0];
}

// ------- pack pw_w [O,C] into per-lane WMMA B fragments (f16) -------
// fragment f = (otile, kstep); element [lane][half i]:
//   o = otile*16 + (lane&15), k = kstep*32 + i + (lane>=16 ? 16 : 0)
__global__ void k_pack(const float* __restrict__ W, _Float16* __restrict__ Wp) {
  int idx = blockIdx.x * blockDim.x + threadIdx.x;   // 0 .. O*C-1
  int f = idx >> 9, r = idx & 511;
  int lane = r >> 4, i = r & 15;
  int otile = f >> 3, ks = f & 7;
  int o = otile * 16 + (lane & 15);
  int k = ks * 32 + i + ((lane & 16) ? 16 : 0);
  Wp[idx] = (_Float16)W[o * C + k];
}

// ------- PLIF head: x[T,B,C,N] f32 -> spikes h[T,B,N,C] f16 (LDS transpose) -------
__global__ __launch_bounds__(256)
void k_plif(const float* __restrict__ x, const float* __restrict__ ws,
            const float* __restrict__ bh, _Float16* __restrict__ h) {
  __shared__ float tile[32][33];
  int tx = threadIdx.x, ty = threadIdx.y;            // block (32, 8)
  int n0 = blockIdx.x * 32, c0 = blockIdx.y * 32, b = blockIdx.z;
  float v[4], wsv[4], bhv[4];
#pragma unroll
  for (int j = 0; j < 4; ++j) {
    int c = c0 + ty + 8 * j;
    v[j] = 0.f; wsv[j] = ws[c]; bhv[j] = bh[c];
  }
  for (int t = 0; t < T; ++t) {
#pragma unroll
    for (int j = 0; j < 4; ++j) {
      int c = c0 + ty + 8 * j;
      float xv = x[(((size_t)t * Bc + b) * C + c) * N + n0 + tx];
      v[j] = v[j] + (xv - v[j]) * wsv[j] + bhv[j];
      float s = (v[j] >= VTH) ? 1.f : 0.f;
      v[j] *= (1.f - s);                              // hard reset (detached)
      tile[ty + 8 * j][tx] = s;
    }
    __syncthreads();
#pragma unroll
    for (int j = 0; j < 4; ++j) {
      int n = n0 + ty + 8 * j;
      h[(((size_t)t * Bc + b) * N + n) * C + c0 + tx] = (_Float16)tile[tx][ty + 8 * j];
    }
    __syncthreads();
  }
}

// ------- WMMA GEMM: p[m, o] = h[m, :] . W[o, :]^T + pw_b ; p stored [T,B,O,N] -------
__global__ __launch_bounds__(256)
void k_gemm(const _Float16* __restrict__ h, const _Float16* __restrict__ Wp,
            const float* __restrict__ pb, float* __restrict__ p) {
  int wave = threadIdx.x >> 5, lane = threadIdx.x & 31;
  int hsel = (lane & 16) ? 1 : 0;
  int mbase = blockIdx.x * 16;
  int otile0 = blockIdx.y * 16 + wave * 2;           // each wave: 2 adjacent o-tiles
  const _Float16* hA = h + (size_t)(mbase + (lane & 15)) * C;
  v8f acc0 = {}, acc1 = {};
#pragma unroll
  for (int ks = 0; ks < 8; ++ks) {
    const int kb = ks * 32 + hsel * 16;
    v8h alo = *(const v8h*)(hA + kb);                // halves 0..7
    v8h ahi = *(const v8h*)(hA + kb + 8);            // halves 8..15
    union { v16h v; v8h h2[2]; } ua;
    ua.h2[0] = alo; ua.h2[1] = ahi;
    v16h b0 = *(const v16h*)(Wp + ((size_t)(otile0 * 8 + ks) * 32 + lane) * 16);
    v16h b1 = *(const v16h*)(Wp + ((size_t)((otile0 + 1) * 8 + ks) * 32 + lane) * 16);
    acc0 = __builtin_amdgcn_wmma_f32_16x16x32_f16(false, ua.v, false, b0, (short)0, acc0, false, false);
    acc1 = __builtin_amdgcn_wmma_f32_16x16x32_f16(false, ua.v, false, b1, (short)0, acc1, false, false);
  }
  int tb    = mbase >> 10;                           // mbase / N (N = 1024)
  int nbase = (mbase & (N - 1)) + hsel * 8;
  int o0 = otile0 * 16 + (lane & 15);
  int o1 = o0 + 16;
  float pb0 = pb[o0], pb1 = pb[o1];
  float* p0 = p + ((size_t)tb * O + o0) * N + nbase;
  float* p1 = p + ((size_t)tb * O + o1) * N + nbase;
#pragma unroll
  for (int r = 0; r < 8; ++r) { p0[r] = acc0[r] + pb0; p1[r] = acc1[r] + pb1; }
}

// ------- BN1: per-channel mean/var over (T,B,N) -> affine scale/shift -------
__global__ __launch_bounds__(256)
void k_bn1(const float* __restrict__ p, const float* __restrict__ g, const float* __restrict__ b,
           float* __restrict__ scale, float* __restrict__ shift) {
  __shared__ float rs[256], rq[256];
  int o = blockIdx.x, tid = threadIdx.x;
  float s = 0.f, q = 0.f;
  for (int tb = 0; tb < T * Bc; ++tb) {
    const float4* row = (const float4*)(p + ((size_t)tb * O + o) * N);
    float4 v = row[tid];
    s += v.x + v.y + v.z + v.w;
    q += v.x * v.x + v.y * v.y + v.z * v.z + v.w * v.w;
  }
  rs[tid] = s; rq[tid] = q; __syncthreads();
  for (int st = 128; st > 0; st >>= 1) {
    if (tid < st) { rs[tid] += rs[tid + st]; rq[tid] += rq[tid + st]; }
    __syncthreads();
  }
  if (tid == 0) {
    float cnt  = (float)(T * Bc * N);
    float mean = rs[0] / cnt;
    float var  = rq[0] / cnt - mean * mean;
    float sc   = g[o] * rsqrtf(var + EPS);
    scale[o] = sc; shift[o] = b[o] - mean * sc;
  }
}

// ------- branches: BN1 affine + LIF over T + depthwise conv (in place) + stats -------
__global__ __launch_bounds__(256)
void k_branch(float* __restrict__ p, const float* __restrict__ scale, const float* __restrict__ shift,
              const float* __restrict__ dw1, const float* __restrict__ dw2, const float* __restrict__ dw3,
              float* __restrict__ bsum, float* __restrict__ bsq) {
  __shared__ float sbuf[N + 8];
  __shared__ float red[512];
  int o = blockIdx.x, b = blockIdx.y, tid = threadIdx.x;
  int branch = o >> 8, oc = o & 255;
  int ksz, pad; float kw[5] = {0, 0, 0, 0, 0};
  if (branch == 0)      { ksz = 1; pad = 0; kw[0] = dw1[oc]; }
  else if (branch == 1) { ksz = 3; pad = 1; for (int j = 0; j < 3; ++j) kw[j] = dw2[oc * 3 + j]; }
  else                  { ksz = 5; pad = 2; for (int j = 0; j < 5; ++j) kw[j] = dw3[oc * 5 + j]; }
  float sc = scale[o], sh = shift[o];
  if (tid < 4) { sbuf[tid] = 0.f; sbuf[4 + N + tid] = 0.f; }     // zero halos once
  float v[4] = {0.f, 0.f, 0.f, 0.f};
  float ls = 0.f, lq = 0.f;
  for (int t = 0; t < T; ++t) {
    size_t base = (((size_t)t * Bc + b) * O + o) * N;
#pragma unroll
    for (int j = 0; j < 4; ++j) {
      int n = tid + 256 * j;
      float pv = p[base + n] * sc + sh;                          // BN1 affine
      v[j] = v[j] * DECAY + pv;                                  // LIF, decay_input=False
      float s = (v[j] >= VTH) ? 1.f : 0.f;
      v[j] *= (1.f - s);
      sbuf[4 + n] = s;
    }
    __syncthreads();
#pragma unroll
    for (int j = 0; j < 4; ++j) {
      int n = tid + 256 * j;
      float acc = 0.f;
      for (int q = 0; q < ksz; ++q) acc += kw[q] * sbuf[4 + n - pad + q];
      p[base + n] = acc;                                         // in-place conv output
      ls += acc; lq += acc * acc;
    }
    __syncthreads();
  }
  red[tid] = ls; red[256 + tid] = lq; __syncthreads();
  for (int st = 128; st > 0; st >>= 1) {
    if (tid < st) { red[tid] += red[tid + st]; red[256 + tid] += red[256 + tid + st]; }
    __syncthreads();
  }
  if (tid == 0) { atomicAdd(&bsum[o], red[0]); atomicAdd(&bsq[o], red[256]); }
}

// ------- finalize bn2/3/4 affines -------
__global__ void k_bn2(const float* __restrict__ bsum, const float* __restrict__ bsq,
                      const float* __restrict__ g2, const float* __restrict__ b2,
                      const float* __restrict__ g3, const float* __restrict__ b3,
                      const float* __restrict__ g4, const float* __restrict__ b4,
                      float* __restrict__ scale, float* __restrict__ shift) {
  int o = blockIdx.x * blockDim.x + threadIdx.x;
  if (o >= O) return;
  int branch = o >> 8, c = o & 255;
  float cnt  = (float)(T * Bc * N);
  float mean = bsum[o] / cnt;
  float var  = bsq[o] / cnt - mean * mean;
  float g  = (branch == 0) ? g2[c] : (branch == 1) ? g3[c] : g4[c];
  float bb = (branch == 0) ? b2[c] : (branch == 1) ? b3[c] : b4[c];
  float sc = g * rsqrtf(var + EPS);
  scale[o] = sc; shift[o] = bb - mean * sc;
}

// ------- sum branches + groups=2 channel shuffle -------
__global__ __launch_bounds__(256)
void k_out(const float* __restrict__ p, const float* __restrict__ scale, const float* __restrict__ shift,
           float* __restrict__ out) {
  int n  = blockIdx.x * 256 + threadIdx.x;
  int cp = blockIdx.y;                               // output channel (post-shuffle)
  int tb = blockIdx.z;
  int cs = (cp & 1) * 128 + (cp >> 1);               // source channel (pre-shuffle)
  float acc = 0.f;
#pragma unroll
  for (int br = 0; br < 3; ++br) {
    int o = br * 256 + cs;
    acc += p[((size_t)tb * O + o) * N + n] * scale[o] + shift[o];
  }
  out[((size_t)tb * C + cp) * N + n] = acc;
}

extern "C" void kernel_launch(void* const* d_in, const int* in_sizes, int n_in,
                              void* d_out, int out_size, void* d_ws, size_t ws_size,
                              hipStream_t stream) {
  (void)in_sizes; (void)n_in; (void)out_size; (void)ws_size;
  const float* x      = (const float*)d_in[0];
  const float* w_head = (const float*)d_in[1];
  const float* b_head = (const float*)d_in[2];
  const float* pw_w   = (const float*)d_in[3];
  const float* pw_b   = (const float*)d_in[4];
  const float* dw1    = (const float*)d_in[5];
  const float* dw2    = (const float*)d_in[6];
  const float* dw3    = (const float*)d_in[7];
  const float* bn1_g  = (const float*)d_in[8];
  const float* bn1_b  = (const float*)d_in[9];
  const float* bn2_g  = (const float*)d_in[10];
  const float* bn2_b  = (const float*)d_in[11];
  const float* bn3_g  = (const float*)d_in[12];
  const float* bn3_b  = (const float*)d_in[13];
  const float* bn4_g  = (const float*)d_in[14];
  const float* bn4_b  = (const float*)d_in[15];

  char* ws = (char*)d_ws;
  float*    wsm  = (float*)(ws + OFF_WS);
  _Float16* Wp   = (_Float16*)(ws + OFF_WPK);
  float*    b1s  = (float*)(ws + OFF_B1S);
  float*    b1h  = (float*)(ws + OFF_B1H);
  float*    bsum = (float*)(ws + OFF_SUM);
  float*    bsq  = (float*)(ws + OFF_SQ);
  float*    b2s  = (float*)(ws + OFF_B2S);
  float*    b2h  = (float*)(ws + OFF_B2H);
  _Float16* h    = (_Float16*)(ws + OFF_H);
  float*    p    = (float*)(ws + OFF_P);
  float*    out  = (float*)d_out;

  k_softmax<<<1, C, 0, stream>>>(w_head, wsm);
  k_pack<<<(O * C) / 256, 256, 0, stream>>>(pw_w, Wp);
  k_plif<<<dim3(N / 32, C / 32, Bc), dim3(32, 8), 0, stream>>>(x, wsm, b_head, h);
  k_gemm<<<dim3(M / 16, O / 256), 256, 0, stream>>>(h, Wp, pw_b, p);
  k_bn1<<<O, 256, 0, stream>>>(p, bn1_g, bn1_b, b1s, b1h);
  (void)hipMemsetAsync(ws + OFF_SUM, 0, 2 * O * sizeof(float), stream);
  k_branch<<<dim3(O, Bc), 256, 0, stream>>>(p, b1s, b1h, dw1, dw2, dw3, bsum, bsq);
  k_bn2<<<3, 256, 0, stream>>>(bsum, bsq, bn2_g, bn2_b, bn3_g, bn3_b, bn4_g, bn4_b, b2s, b2h);
  k_out<<<dim3(N / 256, C, T * Bc), 256, 0, stream>>>(p, b2s, b2h, out);
}
